// VectorQuantizer_42829413876013
// MI455X (gfx1250) — matrix operand order repaired and checked by reading, not compiled, not run on hardware
//
#include <hip/hip_runtime.h>
#include <hip/hip_bf16.h>

// ---------------------------------------------------------------------------
// VQ-VAE vector quantizer for MI455X (gfx1250, wave32, WMMA).
//
// score(row, code) = ||e_code||^2 - 2 * <z_row, e_code>   (row norm cancels in
// the argmin).  Dot products via v_wmma_f32_16x16x32_bf16 with fp32 accum;
// codebook norms computed from the same bf16-rounded values for consistency.
// Codebook is pre-converted to bf16 in workspace; the hot kernel streams it
// through LDS with double-buffered GLOBAL_LOAD_ASYNC_TO_LDS_B128 (ASYNCcnt).
// ---------------------------------------------------------------------------

typedef __attribute__((ext_vector_type(16))) __bf16 v16bf;
typedef __attribute__((ext_vector_type(8)))  __bf16 v8bf;
typedef __attribute__((ext_vector_type(4)))  __bf16 v4bf;
typedef __attribute__((ext_vector_type(8)))  float  v8f;
typedef __attribute__((ext_vector_type(4)))  int    v4i;

#define AS1 __attribute__((address_space(1)))
#define AS3 __attribute__((address_space(3)))

#define DDIM        256
#define KCODES      8192
#define NROWS       32768
#define KCHUNK      64                  // codes per LDS tile
#define NCHUNK      (KCODES / KCHUNK)   // 128
#define ROWS_PER_WG 128                 // 8 waves * 16 rows

#if __has_builtin(__builtin_amdgcn_global_load_async_to_lds_b128) && \
    __has_builtin(__builtin_amdgcn_s_wait_asynccnt)
#define VQ_ASYNC 1
#else
#define VQ_ASYNC 0
#endif

// ---------------------------------------------------------------------------
// Kernel 0: convert codebook fp32 -> bf16 workspace copy (one-time, 4 MB).
// ---------------------------------------------------------------------------
__global__ __launch_bounds__(256) void vq_cb_prep(const float* __restrict__ cb,
                                                  __bf16* __restrict__ cb_bf) {
    int i = (blockIdx.x * 256 + threadIdx.x) * 4;
    float4 v = *(const float4*)(cb + i);
    v4bf o;
    o[0] = (__bf16)v.x; o[1] = (__bf16)v.y;
    o[2] = (__bf16)v.z; o[3] = (__bf16)v.w;
    *(v4bf*)(cb_bf + i) = o;
}

// ---------------------------------------------------------------------------
// Kernel 1: codebook norms of the bf16-rounded codebook. 8192 threads.
// ---------------------------------------------------------------------------
__global__ __launch_bounds__(256) void vq_cb_norms(const float* __restrict__ cb,
                                                   float* __restrict__ en) {
    int code = blockIdx.x * blockDim.x + threadIdx.x;   // 0..8191
    const float* c = cb + (size_t)code * DDIM;
    float s = 0.0f;
    #pragma unroll 4
    for (int i = 0; i < DDIM; i += 4) {
        float4 v = *(const float4*)(c + i);
        float a = (float)(__bf16)v.x;
        float b = (float)(__bf16)v.y;
        float d = (float)(__bf16)v.z;
        float e = (float)(__bf16)v.w;
        s = fmaf(a, a, s); s = fmaf(b, b, s);
        s = fmaf(d, d, s); s = fmaf(e, e, s);
    }
    en[code] = s;
}

// ---------------------------------------------------------------------------
// Kernel 2: fused distance-GEMM + argmin.  256 blocks x 256 threads (8 waves).
// Wave w owns rows [blk*128 + 16w, +16); A fragments live in VGPRs.
// bf16 codebook streamed through double-buffered LDS tiles via async DMA.
// ---------------------------------------------------------------------------
__global__ __launch_bounds__(256) void vq_argmin(const float* __restrict__ z,
                                                 const __bf16* __restrict__ cb_bf,
                                                 const float* __restrict__ en,
                                                 int* __restrict__ out_idx) {
    __shared__ __bf16 sB[2][KCHUNK * DDIM];   // 2 x 32 KB

    const int t    = threadIdx.x;
    const int lane = t & 31;
    const int wave = t >> 5;
    const int n16  = lane & 15;            // column / A-row selector
    const int hi   = lane >> 4;            // K-half selector
    const int r0   = blockIdx.x * ROWS_PER_WG + wave * 16;

    // ---- Load A fragments (16 rows x 256, bf16) into registers.
    // A 16x32 layout (matches B): lanes 0-15 hold K {0..15},
    // lanes 16-31 hold K {16..31}.
    v16bf a[8];
    {
        const float* zr = z + (size_t)(r0 + n16) * DDIM;
        #pragma unroll
        for (int s = 0; s < 8; ++s) {
            const float* p0 = zr + s * 32 + hi * 16;
            const float* p1 = p0 + 8;
            float4 f0 = *(const float4*)(p0);
            float4 f1 = *(const float4*)(p0 + 4);
            float4 f2 = *(const float4*)(p1);
            float4 f3 = *(const float4*)(p1 + 4);
            v16bf av;
            av[0]=(__bf16)f0.x;  av[1]=(__bf16)f0.y;  av[2]=(__bf16)f0.z;  av[3]=(__bf16)f0.w;
            av[4]=(__bf16)f1.x;  av[5]=(__bf16)f1.y;  av[6]=(__bf16)f1.z;  av[7]=(__bf16)f1.w;
            av[8]=(__bf16)f2.x;  av[9]=(__bf16)f2.y;  av[10]=(__bf16)f2.z; av[11]=(__bf16)f2.w;
            av[12]=(__bf16)f3.x; av[13]=(__bf16)f3.y; av[14]=(__bf16)f3.z; av[15]=(__bf16)f3.w;
            a[s] = av;
        }
    }

    float best[8];
    int   bidx[8];
    #pragma unroll
    for (int j = 0; j < 8; ++j) { best[j] = 3.4e38f; bidx[j] = 0; }

    // ---- Tile copy: 32 KB = 2048 x 16 B; 8 x b128 per thread.
    auto issue_copy = [&](int buf, int c0) {
        const __bf16* src = cb_bf + (size_t)c0 * DDIM;
        #pragma unroll
        for (int it = 0; it < 8; ++it) {
            int f = it * 256 + t;                 // 16-byte packet index
#if VQ_ASYNC
            __builtin_amdgcn_global_load_async_to_lds_b128(
                (AS1 v4i*)(void*)(src + f * 8),
                (AS3 v4i*)(void*)(&sB[buf][f * 8]),
                0, 0);
#else
            *(v8bf*)(&sB[buf][f * 8]) = *(const v8bf*)(src + f * 8);
#endif
        }
    };

    issue_copy(0, 0);

    for (int i = 0; i < NCHUNK; ++i) {
        const int cur = i & 1;
#if VQ_ASYNC
        __builtin_amdgcn_s_wait_asynccnt(0);   // our slice of tile i landed
#endif
        __syncthreads();                        // whole tile i visible; tile i-1 reads done
        if (i + 1 < NCHUNK) issue_copy(cur ^ 1, (i + 1) * KCHUNK);  // overlap with compute

        const int c0 = i * KCHUNK;
        #pragma unroll
        for (int ct = 0; ct < KCHUNK / 16; ++ct) {
            // B 32x16 bf16 layout: lane = column n16, half-wave hi picks the
            // contiguous 16-element K block -> one 32-byte LDS read per step.
            const __bf16* bp = &sB[cur][(ct * 16 + n16) * DDIM + hi * 16];
            v16bf bfr[8];
            #pragma unroll
            for (int s = 0; s < 8; ++s)
                bfr[s] = *(const v16bf*)(bp + s * 32);
            v8f acc = {};
            #pragma unroll
            for (int s = 0; s < 8; ++s)
                acc = __builtin_amdgcn_wmma_f32_16x16x32_bf16(
                          false, a[s], false, bfr[s], (short)0, acc, false, false);

            float ev   = en[c0 + ct * 16 + n16];
            int   code = c0 + ct * 16 + n16;
            #pragma unroll
            for (int j = 0; j < 8; ++j) {
                float score = fmaf(-2.0f, acc[j], ev);
                if (score < best[j]) { best[j] = score; bidx[j] = code; }
            }
        }
    }

    // ---- Cross-lane argmin within each 16-lane half (same output row set).
    #pragma unroll
    for (int j = 0; j < 8; ++j) {
        #pragma unroll
        for (int m = 1; m < 16; m <<= 1) {
            float ob = __shfl_xor(best[j], m, 32);
            int   oi = __shfl_xor(bidx[j], m, 32);
            if (ob < best[j] || (ob == best[j] && oi < bidx[j])) {
                best[j] = ob; bidx[j] = oi;
            }
        }
    }
    // C/D layout: VGPR j of half-wave hi holds row r0 + hi*8 + j.
    if (n16 == 0) {
        #pragma unroll
        for (int j = 0; j < 8; ++j)
            out_idx[r0 + hi * 8 + j] = bidx[j];
    }
}

// ---------------------------------------------------------------------------
// Kernel 3: gather z_q, write indices-as-float, deterministic per-block
// squared-error partials (fixed-order LDS tree, no float atomics).
// ---------------------------------------------------------------------------
__global__ __launch_bounds__(256) void vq_gather(const float* __restrict__ z,
                                                 const float* __restrict__ cb,
                                                 const int* __restrict__ idx,
                                                 float* __restrict__ out_zq,
                                                 float* __restrict__ out_idx_f,
                                                 float* __restrict__ partial) {
    __shared__ float red[256];
    const int t    = threadIdx.x;
    const int base = blockIdx.x * ROWS_PER_WG;
    float acc = 0.0f;
    for (int r = 0; r < ROWS_PER_WG; ++r) {
        int   row = base + r;
        int   id  = idx[row];
        float q   = cb[(size_t)id * DDIM + t];
        float zv  = z[(size_t)row * DDIM + t];
        float d   = q - zv;
        acc = fmaf(d, d, acc);
        out_zq[(size_t)row * DDIM + t] = q;
        if (t == 0) out_idx_f[row] = (float)id;
    }
    red[t] = acc;
    __syncthreads();
    for (int s = 128; s > 0; s >>= 1) {
        if (t < s) red[t] += red[t + s];
        __syncthreads();
    }
    if (t == 0) partial[blockIdx.x] = red[0];
}

__global__ __launch_bounds__(256) void vq_finalize(const float* __restrict__ partial,
                                                   float* __restrict__ out_loss) {
    __shared__ float red[256];
    const int t = threadIdx.x;
    red[t] = partial[t];
    __syncthreads();
    for (int s = 128; s > 0; s >>= 1) {
        if (t < s) red[t] += red[t + s];
        __syncthreads();
    }
    // commitment_loss = (1 + BETA) * mean((z_q - z)^2),  BETA = 0.25
    if (t == 0) out_loss[0] = 1.25f * red[0] / (float)((size_t)NROWS * DDIM);
}

// ---------------------------------------------------------------------------
extern "C" void kernel_launch(void* const* d_in, const int* in_sizes, int n_in,
                              void* d_out, int out_size, void* d_ws, size_t ws_size,
                              hipStream_t stream) {
    const float* z  = (const float*)d_in[0];   // (16,2048,256) fp32
    const float* cb = (const float*)d_in[1];   // (8192,256)    fp32

    float* out      = (float*)d_out;
    float* zq_out   = out;                                // 8388608 floats
    float* loss_out = out + (size_t)NROWS * DDIM;         // 1 float
    float* idx_out  = loss_out + 1;                       // 32768 floats

    __bf16* cb_bf   = (__bf16*)d_ws;                      // 2M bf16 (4 MB)
    float*  en      = (float*)(cb_bf + (size_t)KCODES * DDIM);  // 8192 f32
    int*    idx_ws  = (int*)(en + KCODES);                // 32768 i32
    float*  partial = (float*)(idx_ws + NROWS);           // 256 f32

    vq_cb_prep <<<(KCODES * DDIM) / (256 * 4), 256, 0, stream>>>(cb, cb_bf);
    vq_cb_norms<<<KCODES / 256, 256, 0, stream>>>(cb, en);
    vq_argmin  <<<NROWS / ROWS_PER_WG, 256, 0, stream>>>(z, cb_bf, en, idx_ws);
    vq_gather  <<<NROWS / ROWS_PER_WG, 256, 0, stream>>>(z, cb, idx_ws,
                                                         zq_out, idx_out, partial);
    vq_finalize<<<1, 256, 0, stream>>>(partial, loss_out);
}
